// PoseGatEncoder_3496103379476
// MI455X (gfx1250) — matrix operand order repaired and checked by reading, not compiled, not run on hardware
//
#include <hip/hip_runtime.h>

// ---------------------------------------------------------------------------
// PoseGAT encoder for gfx1250 (MI455X): bf16 WMMA for all heavy matmuls.
//  - layers 1/2: per-graph WMMA transforms (A fragment shared across Wl/Wr)
//  - output projection: 32x32 register-blocked bf16 WMMA GEMM (2x2 tiles/wave)
// ---------------------------------------------------------------------------

typedef __attribute__((ext_vector_type(16))) __bf16 v16bf;
typedef __attribute__((ext_vector_type(8)))  __bf16 v8bf;
typedef __attribute__((ext_vector_type(8)))  float  v8f;

#define NJ 50      // joints per graph
#define NE 144     // directed edges incl. self loops (94 + 50)
#define NH 4       // heads (all layers)
#define NG 2048    // graphs = B*T = 16*128

__host__ __device__ constexpr size_t rnd16(size_t x) { return (x + 15) & ~(size_t)15; }

// Dynamic-LDS layout for the WMMA GAT layer kernel.
template<int DIN, int DOUT>
struct LSm {
  static constexpr size_t XL  = 0;                                  // float [NJ][DOUT]
  static constexpr size_t XR  = XL  + rnd16((size_t)NJ * DOUT * 4); // float [NJ][DOUT]
  static constexpr size_t ES  = XR  + rnd16((size_t)NJ * DOUT * 4); // float [NE][NH]
  static constexpr size_t MS  = ES  + rnd16((size_t)NE * NH * 4);   // float [NJ][NH]
  static constexpr size_t DN  = MS  + rnd16((size_t)NJ * NH * 4);   // float [NJ][NH]
  static constexpr size_t SRC = DN  + rnd16((size_t)NJ * NH * 4);   // int [NE]
  static constexpr size_t DST = SRC + rnd16((size_t)NE * 4);        // int [NE]
  static constexpr size_t CNT = DST + rnd16((size_t)NE * 4);        // int [NJ]
  static constexpr size_t IDX = CNT + rnd16((size_t)NJ * 4);        // int [NJ][8] CSR
  static constexpr size_t XB  = IDX + rnd16((size_t)NJ * 8 * 4);    // bf16 [64][DIN]
  static constexpr size_t TOTAL = XB + rnd16((size_t)64 * DIN * 2);
};

// ---------------------------------------------------------------------------
// Weight transpose + fp32 -> bf16:  src[K][N] -> dst[N][K]
// ---------------------------------------------------------------------------
__global__ __launch_bounds__(256) void transpose_bf16(
    const float* __restrict__ src, __bf16* __restrict__ dst, int K, int N)
{
  const int total = K * N;
  for (int i = blockIdx.x * 256 + threadIdx.x; i < total; i += gridDim.x * 256) {
    const int n = i / K, k = i % K;
    dst[i] = (__bf16)src[k * N + n];
  }
}

// ---------------------------------------------------------------------------
// Shared attention tail: edge scores -> per-dst softmax -> aggregate -> ELU.
// xl/xr are f32 [NJ][DOUT] in LDS; eS reused in-place for alpha.
// Deterministic (CSR gather, no float atomics).
// ---------------------------------------------------------------------------
template<int DOUT>
__device__ __forceinline__ void gat_attention(
    int tid, const float* __restrict__ att, const float* __restrict__ bias,
    const float* xl, const float* xr, float* eS, float* mS, float* dnS,
    const int* srcS, const int* dstS, const int* cnt, const int* idx,
    __bf16* __restrict__ og)
{
  constexpr int C = DOUT / NH;
  __syncthreads();
  // e[e][h] = sum_c att[h][c] * leaky_relu(xl[src][h,c] + xr[dst][h,c], 0.2)
  for (int t = tid; t < NE * NH; t += 256) {
    const int e = t >> 2, h = t & 3;
    const float* pl = xl + (size_t)srcS[e] * DOUT + h * C;
    const float* pr = xr + (size_t)dstS[e] * DOUT + h * C;
    const float* pa = att + h * C;
    float s = 0.f;
    for (int c = 0; c < C; ++c) {
      float v = pl[c] + pr[c];
      v = v > 0.f ? v : 0.2f * v;
      s += pa[c] * v;
    }
    eS[t] = s;
  }
  __syncthreads();
  // per-(dst, head) max / sum-of-exp over incoming edges
  for (int t = tid; t < NJ * NH; t += 256) {
    const int j = t >> 2, h = t & 3;
    const int d = cnt[j];
    float mx = -3.0e38f;
    for (int q = 0; q < d; ++q) mx = fmaxf(mx, eS[idx[j * 8 + q] * NH + h]);
    float s = 0.f;
    for (int q = 0; q < d; ++q) s += __expf(eS[idx[j * 8 + q] * NH + h] - mx);
    mS[t] = mx; dnS[t] = s;
  }
  __syncthreads();
  // alpha in place of e
  for (int t = tid; t < NE * NH; t += 256) {
    const int e = t >> 2, h = t & 3;
    const int d = dstS[e];
    eS[t] = __expf(eS[t] - mS[d * NH + h]) / dnS[d * NH + h];
  }
  __syncthreads();
  // out[j][oc] = ELU( sum_{e: dst==j} alpha[e][h] * xl[src[e]][oc] + bias[oc] )
  for (int t = tid; t < NJ * DOUT; t += 256) {
    const int j = t / DOUT, oc = t % DOUT;
    const int h = oc / C;
    const int d = cnt[j];
    float s = 0.f;
    for (int q = 0; q < d; ++q) {
      const int e = idx[j * 8 + q];
      s += eS[e * NH + h] * xl[(size_t)srcS[e] * DOUT + oc];
    }
    float v = s + bias[oc];
    v = v > 0.f ? v : __expf(v) - 1.f;
    og[t] = (__bf16)v;
  }
}

// ---------------------------------------------------------------------------
// Layer 0: din=3 (too small for K=32 WMMA) -> VALU transforms + attention.
// One workgroup per graph.
// ---------------------------------------------------------------------------
__global__ __launch_bounds__(256) void gat_layer0(
    const float* __restrict__ x_seq,
    const float* __restrict__ Wl, const float* __restrict__ bl,
    const float* __restrict__ Wr, const float* __restrict__ br,
    const float* __restrict__ att, const float* __restrict__ bias,
    const int* __restrict__ srcE, const int* __restrict__ dstE,
    __bf16* __restrict__ xout)
{
  constexpr int DOUT = 64;
  __shared__ float x0[NJ * 3];
  __shared__ float xl[NJ * DOUT];
  __shared__ float xr[NJ * DOUT];
  __shared__ float eS[NE * NH];
  __shared__ float mS[NJ * NH];
  __shared__ float dnS[NJ * NH];
  __shared__ int srcS[NE], dstS[NE], cnt[NJ], idx[NJ * 8];

  const int tid = threadIdx.x, g = blockIdx.x;
  const float* xg = x_seq + (size_t)g * (NJ * 3);
  for (int i = tid; i < NJ * 3; i += 256) x0[i] = xg[i];
  for (int i = tid; i < NE; i += 256) { srcS[i] = srcE[i]; dstS[i] = dstE[i]; }
  __syncthreads();

  for (int t = tid; t < NJ * DOUT; t += 256) {
    const int j = t / DOUT, o = t % DOUT;
    const float a0 = x0[j * 3 + 0], a1 = x0[j * 3 + 1], a2 = x0[j * 3 + 2];
    xl[t] = bl[o] + a0 * Wl[o] + a1 * Wl[DOUT + o] + a2 * Wl[2 * DOUT + o];
    xr[t] = br[o] + a0 * Wr[o] + a1 * Wr[DOUT + o] + a2 * Wr[2 * DOUT + o];
  }
  for (int j = tid; j < NJ; j += 256) {
    int c = 0;
    for (int e = 0; e < NE; ++e) if (dstS[e] == j) idx[j * 8 + c++] = e;
    cnt[j] = c;
  }
  gat_attention<DOUT>(tid, att, bias, xl, xr, eS, mS, dnS, srcS, dstS, cnt, idx,
                      xout + (size_t)g * (NJ * DOUT));
}

// ---------------------------------------------------------------------------
// Generic GATv2 layer with bf16 WMMA transforms. One workgroup per graph,
// 8 waves; each wave computes a 16x16 tile of BOTH xl and xr per step,
// reusing one A fragment for two v_wmma_f32_16x16x32_bf16 issues.
// ---------------------------------------------------------------------------
template<int DIN, int DOUT>
__global__ __launch_bounds__(256) void gat_layer_wmma(
    const __bf16* __restrict__ xin,
    const __bf16* __restrict__ WlT, const float* __restrict__ bl,
    const __bf16* __restrict__ WrT, const float* __restrict__ br,
    const float* __restrict__ att, const float* __restrict__ bias,
    const int* __restrict__ srcE, const int* __restrict__ dstE,
    __bf16* __restrict__ xout)
{
  using S = LSm<DIN, DOUT>;
  extern __shared__ char smem[];
  float*  xl   = (float*)(smem + S::XL);
  float*  xr   = (float*)(smem + S::XR);
  float*  eS   = (float*)(smem + S::ES);
  float*  mS   = (float*)(smem + S::MS);
  float*  dnS  = (float*)(smem + S::DN);
  int*    srcS = (int*)(smem + S::SRC);
  int*    dstS = (int*)(smem + S::DST);
  int*    cnt  = (int*)(smem + S::CNT);
  int*    idx  = (int*)(smem + S::IDX);
  __bf16* xb   = (__bf16*)(smem + S::XB);

  const int tid = threadIdx.x, g = blockIdx.x;
  const __bf16* xg = xin + (size_t)g * (NJ * DIN);
  for (int i = tid; i < 64 * DIN; i += 256)
    xb[i] = (i < NJ * DIN) ? xg[i] : (__bf16)0.0f;
  for (int i = tid; i < NE; i += 256) { srcS[i] = srcE[i]; dstS[i] = dstE[i]; }
  __syncthreads();

  // ---- WMMA transforms: xl = x @ Wl + bl, xr = x @ Wr + br ----
  const int lane = tid & 31, wave = tid >> 5;
  const int lm = lane & 15, half = lane >> 4;
  constexpr int NTL = DOUT / 16;  // N tiles
  constexpr int KT  = DIN / 32;   // K steps
  for (int t = wave; t < 4 * NTL; t += 8) {   // uniform trip count per wave
    const int mt = t / NTL, nt = t % NTL;
    const int bn = nt * 16 + lm;
    // A fragment: lane m = lm, K run = half*16 .. half*16+15 (contiguous)
    const __bf16* arowp = xb + (size_t)(mt * 16 + lm) * DIN + half * 16;
    // B^T fragments: lane n = bn, K run = half*16 .. half*16+15 (contiguous)
    const __bf16* blp = WlT + (size_t)bn * DIN + half * 16;
    const __bf16* brp = WrT + (size_t)bn * DIN + half * 16;
    v8f accL = {0.f, 0.f, 0.f, 0.f, 0.f, 0.f, 0.f, 0.f};
    v8f accR = {0.f, 0.f, 0.f, 0.f, 0.f, 0.f, 0.f, 0.f};
#pragma unroll
    for (int kt = 0; kt < KT; ++kt) {
      union { v16bf v; v8bf h[2]; } a, wl, wr;
      a.h[0]  = *(const v8bf*)(arowp + kt * 32);
      a.h[1]  = *(const v8bf*)(arowp + kt * 32 + 8);
      wl.h[0] = *(const v8bf*)(blp + kt * 32);
      wl.h[1] = *(const v8bf*)(blp + kt * 32 + 8);
      wr.h[0] = *(const v8bf*)(brp + kt * 32);
      wr.h[1] = *(const v8bf*)(brp + kt * 32 + 8);
      accL = __builtin_amdgcn_wmma_f32_16x16x32_bf16(
          false, a.v, false, wl.v, (short)0, accL, false, false);
      accR = __builtin_amdgcn_wmma_f32_16x16x32_bf16(
          false, a.v, false, wr.v, (short)0, accR, false, false);
    }
    const float bbl = bl[bn], bbr = br[bn];
#pragma unroll
    for (int v = 0; v < 8; ++v) {   // C/D: VGPR v -> row v (lanes 0-15) / v+8 (16-31)
      const int row = mt * 16 + v + 8 * half;
      if (row < NJ) {
        xl[(size_t)row * DOUT + bn] = accL[v] + bbl;
        xr[(size_t)row * DOUT + bn] = accR[v] + bbr;
      }
    }
  }
  // Deterministic incoming-edge CSR per dst
  for (int j = tid; j < NJ; j += 256) {
    int c = 0;
    for (int e = 0; e < NE; ++e) if (dstS[e] == j) idx[j * 8 + c++] = e;
    cnt[j] = c;
  }
  gat_attention<DOUT>(tid, att, bias, xl, xr, eS, mS, dnS, srcS, dstS, cnt, idx,
                      xout + (size_t)g * (NJ * DOUT));
}

// ---------------------------------------------------------------------------
// Output projection: [2048 x 12800] @ [12800 x 512] + bout, bf16 WMMA, f32 out.
// 2x2 register-blocked: each wave owns a 32x32 output block (4 accumulators),
// 4 WMMAs per 8 fragment loads -> 16 FLOP/byte from L2. 1024 wave tiles,
// 8 waves per block -> 128 blocks, 400 K-steps each.
// ---------------------------------------------------------------------------
__global__ __launch_bounds__(256) void out_gemm(
    const __bf16* __restrict__ A,    // [2048][12800] bf16 (x3 activations)
    const __bf16* __restrict__ Bt,   // [512][12800]  bf16 (Wout^T)
    const float* __restrict__ bout,  // [512]
    float* __restrict__ out)         // [2048][512]
{
  constexpr int K = 12800, N = 512;
  const int wave = threadIdx.x >> 5, lane = threadIdx.x & 31;
  const int lm = lane & 15, half = lane >> 4;
  const int tile = blockIdx.x * 8 + wave;   // 0..1023
  const int mt = tile >> 4;                 // 64 M blocks of 32
  const int nt = tile & 15;                 // 16 N blocks of 32
  const __bf16* arow0 = A  + (size_t)(mt * 32 + lm) * K + half * 16;
  const __bf16* arow1 = arow0 + (size_t)16 * K;
  const __bf16* brow0 = Bt + (size_t)(nt * 32 + lm) * K + half * 16;
  const __bf16* brow1 = brow0 + (size_t)16 * K;
  v8f acc00 = {0.f, 0.f, 0.f, 0.f, 0.f, 0.f, 0.f, 0.f};
  v8f acc01 = acc00, acc10 = acc00, acc11 = acc00;
#pragma unroll 2
  for (int kb = 0; kb < K; kb += 32) {
    union { v16bf v; v8bf h[2]; } a0, a1, b0, b1;
    a0.h[0] = *(const v8bf*)(arow0 + kb);
    a0.h[1] = *(const v8bf*)(arow0 + kb + 8);
    a1.h[0] = *(const v8bf*)(arow1 + kb);
    a1.h[1] = *(const v8bf*)(arow1 + kb + 8);
    b0.h[0] = *(const v8bf*)(brow0 + kb);
    b0.h[1] = *(const v8bf*)(brow0 + kb + 8);
    b1.h[0] = *(const v8bf*)(brow1 + kb);
    b1.h[1] = *(const v8bf*)(brow1 + kb + 8);
    __builtin_prefetch((const void*)(arow0 + kb + 1024), 0, 1); // global_prefetch_b8
    __builtin_prefetch((const void*)(arow1 + kb + 1024), 0, 1);
    __builtin_prefetch((const void*)(brow0 + kb + 1024), 0, 1);
    __builtin_prefetch((const void*)(brow1 + kb + 1024), 0, 1);
    acc00 = __builtin_amdgcn_wmma_f32_16x16x32_bf16(
        false, a0.v, false, b0.v, (short)0, acc00, false, false);
    acc01 = __builtin_amdgcn_wmma_f32_16x16x32_bf16(
        false, a0.v, false, b1.v, (short)0, acc01, false, false);
    acc10 = __builtin_amdgcn_wmma_f32_16x16x32_bf16(
        false, a1.v, false, b0.v, (short)0, acc10, false, false);
    acc11 = __builtin_amdgcn_wmma_f32_16x16x32_bf16(
        false, a1.v, false, b1.v, (short)0, acc11, false, false);
  }
  const int col0 = nt * 32 + lm, col1 = col0 + 16;
  const float bb0 = bout[col0], bb1 = bout[col1];
#pragma unroll
  for (int v = 0; v < 8; ++v) {
    const int row0 = mt * 32 + v + 8 * half;
    const int row1 = row0 + 16;
    out[(size_t)row0 * N + col0] = acc00[v] + bb0;
    out[(size_t)row0 * N + col1] = acc01[v] + bb1;
    out[(size_t)row1 * N + col0] = acc10[v] + bb0;
    out[(size_t)row1 * N + col1] = acc11[v] + bb1;
  }
}

// ---------------------------------------------------------------------------
extern "C" void kernel_launch(void* const* d_in, const int* in_sizes, int n_in,
                              void* d_out, int out_size, void* d_ws, size_t ws_size,
                              hipStream_t stream) {
  (void)in_sizes; (void)n_in; (void)out_size; (void)ws_size;
  const float* x_seq = (const float*)d_in[0];
  const float* Wl0 = (const float*)d_in[1],  *bl0 = (const float*)d_in[2];
  const float* Wr0 = (const float*)d_in[3],  *br0 = (const float*)d_in[4];
  const float* att0 = (const float*)d_in[5], *bias0 = (const float*)d_in[6];
  const float* Wl1 = (const float*)d_in[7],  *bl1 = (const float*)d_in[8];
  const float* Wr1 = (const float*)d_in[9],  *br1 = (const float*)d_in[10];
  const float* att1 = (const float*)d_in[11], *bias1 = (const float*)d_in[12];
  const float* Wl2 = (const float*)d_in[13], *bl2 = (const float*)d_in[14];
  const float* Wr2 = (const float*)d_in[15], *br2 = (const float*)d_in[16];
  const float* att2 = (const float*)d_in[17], *bias2 = (const float*)d_in[18];
  const float* Wout = (const float*)d_in[19], *bout = (const float*)d_in[20];
  const int* src = (const int*)d_in[21];
  const int* dst = (const int*)d_in[22];
  float* out = (float*)d_out;

  // Workspace carve (all bf16 intermediates; ~105 MB total)
  char* w = (char*)d_ws;
  size_t off = 0;
  auto take = [&](size_t bytes) -> char* {
    char* p = w + off;
    off += (bytes + 255) & ~(size_t)255;
    return p;
  };
  __bf16* xb1   = (__bf16*)take((size_t)NG * NJ * 64  * 2);
  __bf16* xb2   = (__bf16*)take((size_t)NG * NJ * 128 * 2);
  __bf16* xb3   = (__bf16*)take((size_t)NG * NJ * 256 * 2);   // = GEMM A [2048][12800]
  __bf16* Wl1T  = (__bf16*)take((size_t)128 * 64 * 2);
  __bf16* Wr1T  = (__bf16*)take((size_t)128 * 64 * 2);
  __bf16* Wl2T  = (__bf16*)take((size_t)256 * 128 * 2);
  __bf16* Wr2T  = (__bf16*)take((size_t)256 * 128 * 2);
  __bf16* WoutT = (__bf16*)take((size_t)512 * 12800 * 2);

  // Weight prep: transpose to [N][K] bf16 so B fragments are contiguous K-runs
  transpose_bf16<<<(64 * 128 + 255) / 256, 256, 0, stream>>>(Wl1, Wl1T, 64, 128);
  transpose_bf16<<<(64 * 128 + 255) / 256, 256, 0, stream>>>(Wr1, Wr1T, 64, 128);
  transpose_bf16<<<(128 * 256 + 255) / 256, 256, 0, stream>>>(Wl2, Wl2T, 128, 256);
  transpose_bf16<<<(128 * 256 + 255) / 256, 256, 0, stream>>>(Wr2, Wr2T, 128, 256);
  transpose_bf16<<<(12800 * 512 + 255) / 256, 256, 0, stream>>>(Wout, WoutT, 12800, 512);

  // Layer 0 (K=3: VALU), layers 1-2 (WMMA), output projection (WMMA)
  gat_layer0<<<NG, 256, 0, stream>>>(x_seq, Wl0, bl0, Wr0, br0, att0, bias0,
                                     src, dst, xb1);
  gat_layer_wmma<64, 128><<<NG, 256, (int)LSm<64, 128>::TOTAL, stream>>>(
      xb1, Wl1T, bl1, Wr1T, br1, att1, bias1, src, dst, xb2);
  gat_layer_wmma<128, 256><<<NG, 256, (int)LSm<128, 256>::TOTAL, stream>>>(
      xb2, Wl2T, bl2, Wr2T, br2, att2, bias2, src, dst, xb3);
  out_gemm<<<128, 256, 0, stream>>>(xb3, WoutT, bout, out);
}
